// VQVAE_54812372631699
// MI455X (gfx1250) — compile-verified, matrix operand order checked by reading, not run on hardware
//
#include <hip/hip_runtime.h>
#include <hip/hip_bf16.h>

// ---------------------------------------------------------------------------
// VQ-VAE forward for gfx1250 (MI455X). All heavy math goes through
// v_wmma_f32_16x16x32_bf16 (bf16 multiplicands, fp32 accumulate).
// Convs / deconvs are implicit GEMMs with on-the-fly im2col gathers into LDS.
// Staging is batched (loads -> regs -> LDS) to maximize memory-level
// parallelism; all spatial decompositions are shift/mask (pow2 dims).
// ---------------------------------------------------------------------------

typedef __attribute__((ext_vector_type(16))) __bf16 v16bf;
typedef __attribute__((ext_vector_type(8)))  float  v8f;

__device__ __forceinline__ unsigned short f2bfu(float f) {
  unsigned u = __builtin_bit_cast(unsigned, f);
  unsigned r = u + 0x7fffu + ((u >> 16) & 1u);   // round-to-nearest-even
  return (unsigned short)(r >> 16);
}

// Build a 16x32 bf16 A/B fragment per the CDNA5 16-bit operand layout:
// lanes 0-15 hold K {0..7,16..23}, lanes 16-31 hold K {8..15,24..31}.
__device__ __forceinline__ v16bf load_frag(const unsigned short* rowp, int half) {
  v16bf v;
  const unsigned short* p0 = rowp + half * 8;
#pragma unroll
  for (int i = 0; i < 8; i++) v[i]     = __builtin_bit_cast(__bf16, p0[i]);
#pragma unroll
  for (int i = 0; i < 8; i++) v[i + 8] = __builtin_bit_cast(__bf16, p0[16 + i]);
  return v;
}

__device__ __forceinline__ v8f wmma_bf16(v16bf a, v16bf b, v8f c) {
  return __builtin_amdgcn_wmma_f32_16x16x32_bf16(false, a, false, b, (short)0, c,
                                                 false, false);
}

// ---------------------------------------------------------------------------
// Generic implicit-GEMM parameters
// ---------------------------------------------------------------------------
struct GP {
  const float* __restrict__ in;     // activations, NCHW
  const float* __restrict__ wgt;    // conv: (OC,IC,KH,KW); deconv: (IC,OC,4,4)
  const float* __restrict__ bias;   // (OC)
  const float* __restrict__ resid;  // residual (same layout as out) or null
  float* __restrict__ out;          // NCHW
  const float* __restrict__ xref;   // for reconstruction loss (deconv2)
  float* __restrict__ lpart;        // per-block loss partials
  int IC, IH, IW, OC, OH, OW, KH, KW, S, P;
  int M, K;       // M = tokens (per parity class for deconv); M % BM == 0
  int sOW, sOHW;  // log2(OW), log2(OH*OW)  (output dims are powers of two)
};

// ---------------------------------------------------------------------------
// Tiled WMMA implicit-GEMM. WM waves along M, WN along N, NT 16-wide C tiles
// per wave. Block = WM*WN waves (wave32).
//   Config A (convs/deconv1): WM=4, WN=2, NT=2 -> BM=64, BN=64, 256 threads
//   Config B (deconv2, N=3 ): WM=8, WN=1, NT=1 -> BM=128, BN=16, 256 threads
// MODE 0: direct conv.  MODE 1: ConvTranspose2d k=4,s=2,p=1 split into 4
// output parity classes (py,px); K = IC*2*2, kh = (1-py)+2i, ih = oh'+py-i.
// ---------------------------------------------------------------------------
template <int MODE, int RELU, int RES, int LOSS, int WM, int WN, int NT>
__global__ __launch_bounds__(WM* WN * 32) void gemm_conv(GP p) {
  constexpr int BM   = WM * 16;
  constexpr int BN   = WN * NT * 16;
  constexpr int BK   = 32;
  constexpr int BKP  = BK + 8;            // LDS bank-conflict padding
  constexpr int NTH  = WM * WN * 32;
  constexpr int ROWS = NTH / BK;          // tile rows staged per iteration (8)
  constexpr int AIT  = BM / ROWS;         // A elements per thread per K-step
  constexpr int BIT  = BN / ROWS;         // B elements per thread (MODE1)
  constexpr int B4   = (BN * BK) / 4;     // float4 copies (MODE0)
  constexpr int B4IT = (B4 + NTH - 1) / NTH;

  __shared__ unsigned short As[BM * BKP];
  __shared__ unsigned short Bs[BN * BKP];
  __shared__ float red[NTH];

  const int tid  = threadIdx.x;
  const int lane = tid & 31;
  const int wave = tid >> 5;
  const int wm   = wave % WM;
  const int wn   = wave / WM;
  const int m0   = blockIdx.x * BM;
  const int n0   = blockIdx.y * BN;
  const int py   = (int)(blockIdx.z >> 1);
  const int px   = (int)(blockIdx.z & 1);

  // thread-constant staging coordinates: kk fixed, rows step by ROWS
  const int kk = tid & (BK - 1);
  const int r0 = tid >> 5;

  const int sOW = p.sOW, sOHW = p.sOHW;
  const int mOW = (1 << sOW) - 1, mOHW = (1 << sOHW) - 1;
  const int sOWq = sOW - 1, sQ = sOHW - 2;            // deconv quarter-grids
  const int mOWq = (1 << sOWq) - 1, mQ = (1 << sQ) - 1;

  v8f acc[NT];
#pragma unroll
  for (int t = 0; t < NT; t++) acc[t] = (v8f){0.f, 0.f, 0.f, 0.f, 0.f, 0.f, 0.f, 0.f};

  const int half = (lane >> 4) & 1;
  const int row  = lane & 15;
  const int ksteps = (p.K + BK - 1) / BK;

  for (int ks = 0; ks < ksteps; ks++) {
    const int k0 = ks * BK;

    // ---- hoisted per-K-step k decomposition (kk is thread-constant) ----
    const int kA = k0 + kk;
    const bool kok = kA < p.K;
    int icA, khA, kwA;
    if constexpr (MODE == 0) {
      const int khw = p.KH * p.KW;
      icA = kA / khw;
      const int r = kA - icA * khw;
      khA = r / p.KW;
      kwA = r - khA * p.KW;
    } else {
      icA = kA >> 2;
      const int tap = kA & 3;
      khA = tap >> 1;     // ti
      kwA = tap & 1;      // tj
    }

    __syncthreads();  // previous iteration's frag reads done

    // ---- stage A: batch loads into regs, then LDS stores ----
    float av[AIT];
#pragma unroll
    for (int i = 0; i < AIT; i++) {
      const int m = m0 + r0 + i * ROWS;
      float v = 0.f;
      if constexpr (MODE == 0) {
        const int nimg = m >> sOHW, s = m & mOHW;
        const int oh = s >> sOW, ow = s & mOW;
        const int ih = oh * p.S - p.P + khA;
        const int iw = ow * p.S - p.P + kwA;
        if (kok && (unsigned)ih < (unsigned)p.IH && (unsigned)iw < (unsigned)p.IW)
          v = p.in[((nimg * p.IC + icA) * p.IH + ih) * p.IW + iw];
      } else {
        const int nimg = m >> sQ, s = m & mQ;
        const int ohq = s >> sOWq, owq = s & mOWq;
        const int ih = ohq + py - khA;
        const int iw = owq + px - kwA;
        if (kok && (unsigned)ih < (unsigned)p.IH && (unsigned)iw < (unsigned)p.IW)
          v = p.in[((nimg * p.IC + icA) * p.IH + ih) * p.IW + iw];
      }
      av[i] = v;
    }
#pragma unroll
    for (int i = 0; i < AIT; i++) As[(r0 + i * ROWS) * BKP + kk] = f2bfu(av[i]);

    // ---- stage B ----
    if constexpr (MODE == 0) {
      // conv weights are K-contiguous: B[k][n] = wgt[n*K + k] -> float4 copy
#pragma unroll
      for (int j = 0; j < B4IT; j++) {
        const int f = tid + j * NTH;
        if (f < B4) {
          const int ni = f >> 3;            // (f*4)/BK
          const int kb = (f & 7) * 4;
          const int kg = k0 + kb;
          const float* src = p.wgt + (n0 + ni) * p.K + kg;
          float4 v = make_float4(0.f, 0.f, 0.f, 0.f);
          if (kg + 3 < p.K) {
            v = *(const float4*)src;
          } else {
            if (kg + 0 < p.K) v.x = src[0];
            if (kg + 1 < p.K) v.y = src[1];
            if (kg + 2 < p.K) v.z = src[2];
          }
          unsigned short* d = &Bs[ni * BKP + kb];
          d[0] = f2bfu(v.x); d[1] = f2bfu(v.y); d[2] = f2bfu(v.z); d[3] = f2bfu(v.w);
        }
      }
      // prefetch next weight K-tile (global_prefetch_b8)
      if (tid < BN && k0 + BK < p.K)
        __builtin_prefetch(p.wgt + (n0 + tid) * p.K + k0 + BK, 0, 1);
    } else {
      // deconv weight gather (IC,OC,4,4): batched scalar loads
      const int kh2 = (1 - py) + 2 * khA;
      const int kw2 = (1 - px) + 2 * kwA;
      float bv[BIT];
#pragma unroll
      for (int i = 0; i < BIT; i++) {
        const int n = n0 + r0 + i * ROWS;
        float v = 0.f;
        if (kok && n < p.OC) v = p.wgt[((icA * p.OC + n) * 4 + kh2) * 4 + kw2];
        bv[i] = v;
      }
#pragma unroll
      for (int i = 0; i < BIT; i++) Bs[(r0 + i * ROWS) * BKP + kk] = f2bfu(bv[i]);
    }
    __syncthreads();

    // ---- fragments + WMMA ----
    v16bf af = load_frag(&As[(wm * 16 + row) * BKP], half);
#pragma unroll
    for (int t = 0; t < NT; t++) {
      v16bf bf = load_frag(&Bs[((wn * NT + t) * 16 + row) * BKP], half);
      acc[t] = wmma_bf16(af, bf, acc[t]);
    }
  }

  // ---- epilogue: C vgpr r -> (m = base + r + 8*half, n = ... + (lane&15)) ----
  float lsum = 0.f;
  const int mbase = m0 + wm * 16 + half * 8;
#pragma unroll
  for (int r = 0; r < 8; r++) {
    const int m = mbase + r;
    int nimg, oh, ow;
    if constexpr (MODE == 0) {
      nimg = m >> sOHW;
      const int s = m & mOHW;
      oh = s >> sOW; ow = s & mOW;
    } else {
      nimg = m >> sQ;
      const int s = m & mQ;
      oh = 2 * (s >> sOWq) + py;
      ow = 2 * (s & mOWq) + px;
    }
    const int obase = (oh << sOW) + ow;
#pragma unroll
    for (int t = 0; t < NT; t++) {
      const int n = n0 + (wn * NT + t) * 16 + row;
      if (n < p.OC) {
        float v = acc[t][r] + p.bias[n];
        if constexpr (RELU) v = v > 0.f ? v : 0.f;
        const int oi = ((nimg * p.OC + n) << sOHW) + obase;
        if constexpr (RES) v += p.resid[oi];
        p.out[oi] = v;
        if constexpr (LOSS) { const float d = v - p.xref[oi]; lsum += d * d; }
      }
    }
  }
  if constexpr (LOSS) {
    red[tid] = lsum;
    __syncthreads();
    for (int s = NTH / 2; s > 0; s >>= 1) {
      if (tid < s) red[tid] += red[tid + s];
      __syncthreads();
    }
    if (tid == 0) p.lpart[blockIdx.x + gridDim.x * blockIdx.z] = red[0];
  }
}

// ---------------------------------------------------------------------------
// RMSNorm over (C,H,W)=(256,8,8) per image; weight is a flat 16384 slab that
// indexes identically to the per-image activation slab. float4 data movement.
// ---------------------------------------------------------------------------
__global__ __launch_bounds__(256) void rmsnorm_kernel(const float* __restrict__ in,
                                                      const float* __restrict__ w,
                                                      float* __restrict__ out) {
  __shared__ float red[256];
  const int n = blockIdx.x;
  const float4* src = (const float4*)(in + n * 16384);
  const float4* wv  = (const float4*)w;
  float4* dst = (float4*)(out + n * 16384);
  float s = 0.f;
#pragma unroll 4
  for (int i = threadIdx.x; i < 4096; i += 256) {
    float4 v = src[i];
    s += v.x * v.x + v.y * v.y + v.z * v.z + v.w * v.w;
  }
  red[threadIdx.x] = s;
  __syncthreads();
  for (int st = 128; st > 0; st >>= 1) {
    if (threadIdx.x < st) red[threadIdx.x] += red[threadIdx.x + st];
    __syncthreads();
  }
  const float scale = rsqrtf(red[0] * (1.f / 16384.f) + 1.1920929e-07f);
#pragma unroll 4
  for (int i = threadIdx.x; i < 4096; i += 256) {
    float4 v = src[i], g = wv[i];
    v.x *= scale * g.x; v.y *= scale * g.y; v.z *= scale * g.z; v.w *= scale * g.w;
    dst[i] = v;
  }
}

// ---------------------------------------------------------------------------
// ||e||^2 per codebook row (512 x 256)
// ---------------------------------------------------------------------------
__global__ __launch_bounds__(256) void enorm_kernel(const float* __restrict__ emb,
                                                    float* __restrict__ out) {
  __shared__ float red[256];
  const float* r = emb + blockIdx.x * 256;
  const float v = r[threadIdx.x];
  red[threadIdx.x] = v * v;
  __syncthreads();
  for (int st = 128; st > 0; st >>= 1) {
    if (threadIdx.x < st) red[threadIdx.x] += red[threadIdx.x + st];
    __syncthreads();
  }
  if (threadIdx.x == 0) out[blockIdx.x] = red[0];
}

// ---------------------------------------------------------------------------
// Fused VQ: 32 tokens/block. WMMA z·e^T over all 512 codes; argmin via packed
// (ordered_float << 9 | idx) keys (ties -> lowest idx); then q gather and
// sum((enc-q)^2) partial. Forward-value: oup == q, dict == commit loss.
// ---------------------------------------------------------------------------
__global__ __launch_bounds__(256) void vq_kernel(const float* __restrict__ enc,
                                                 const float* __restrict__ emb,
                                                 const float* __restrict__ enorm,
                                                 float* __restrict__ qout,
                                                 float* __restrict__ lpart) {
  constexpr int BM = 32, K = 256, NE = 512, KP = 264, BKP = 40;
  __shared__ unsigned short As[BM * KP];        // 16.9 KB, staged once
  __shared__ unsigned short Bs[64 * BKP];       // 5.1 KB, per (nc,k0) chunk
  __shared__ unsigned long long best[BM * 64];  // 16 KB
  __shared__ unsigned long long red2[BM * 8];
  __shared__ int sidx[BM];
  __shared__ float redf[256];

  const int tid = threadIdx.x, lane = tid & 31, wave = tid >> 5;
  const int wm = wave & 1;        // 2 waves along tokens
  const int wn = wave >> 1;       // 4 waves along codes
  const int t0 = blockIdx.x * BM;
  const int half = (lane >> 4) & 1, row = lane & 15;

  // ---- stage z tile, token-fastest for coalescing (32 lanes -> 32 consec) ----
  {
    const int mi = tid & 31;         // token within block
    const int kb0 = tid >> 5;        // k base 0..7, step 8
    const int tg = t0 + mi;
    const int nimg = tg >> 6, s = tg & 63;
    const float* abase = enc + nimg * 16384 + s;
#pragma unroll
    for (int c = 0; c < 4; c++) {
      float av[8];
#pragma unroll
      for (int i = 0; i < 8; i++) av[i] = abase[(kb0 + (c * 8 + i) * 8) * 64];
#pragma unroll
      for (int i = 0; i < 8; i++)
        As[mi * KP + kb0 + (c * 8 + i) * 8] = f2bfu(av[i]);
    }
  }

  unsigned long long mybest[8];
#pragma unroll
  for (int r = 0; r < 8; r++) mybest[r] = ~0ull;

  for (int nc = 0; nc < NE; nc += 64) {
    v8f acc = (v8f){0.f, 0.f, 0.f, 0.f, 0.f, 0.f, 0.f, 0.f};
    for (int k0 = 0; k0 < K; k0 += 32) {
      __syncthreads();
      // codebook rows are contiguous: float4 copy (2 per thread)
#pragma unroll
      for (int j = 0; j < 2; j++) {
        const int f = tid + j * 256;       // f < 512
        const int ni = f >> 3;
        const int kb = (f & 7) * 4;
        const float4 v = *(const float4*)(emb + (nc + ni) * 256 + k0 + kb);
        unsigned short* d = &Bs[ni * BKP + kb];
        d[0] = f2bfu(v.x); d[1] = f2bfu(v.y); d[2] = f2bfu(v.z); d[3] = f2bfu(v.w);
      }
      __syncthreads();
      v16bf af = load_frag(&As[(wm * 16 + row) * KP + k0], half);
      v16bf bf = load_frag(&Bs[(wn * 16 + row) * BKP], half);
      acc = wmma_bf16(af, bf, acc);
    }
    const int n = nc + wn * 16 + row;
    const float en = enorm[n];
#pragma unroll
    for (int r = 0; r < 8; r++) {
      const float score = en - 2.f * acc[r];
      unsigned u = __builtin_bit_cast(unsigned, score);
      u = (u & 0x80000000u) ? ~u : (u | 0x80000000u);
      const unsigned long long key = ((unsigned long long)u << 9) | (unsigned)n;
      if (key < mybest[r]) mybest[r] = key;
    }
  }
  const int tokbase = wm * 16 + half * 8;
  const int slot = wn * 16 + row;
#pragma unroll
  for (int r = 0; r < 8; r++) best[(tokbase + r) * 64 + slot] = mybest[r];
  __syncthreads();

  {  // two-stage deterministic argmin reduce
    const int token = tid >> 3, j = tid & 7;
    unsigned long long b = ~0ull;
#pragma unroll
    for (int s = 0; s < 8; s++) {
      const unsigned long long v = best[token * 64 + j * 8 + s];
      if (v < b) b = v;
    }
    red2[token * 8 + j] = b;
  }
  __syncthreads();
  if (tid < BM) {
    unsigned long long b = ~0ull;
#pragma unroll
    for (int j = 0; j < 8; j++) {
      const unsigned long long v = red2[tid * 8 + j];
      if (v < b) b = v;
    }
    sidx[tid] = (int)(b & 511ull);
  }
  __syncthreads();

  // q = embedding[idx]; accumulate (enc - q)^2
  {
    const int token = tid >> 3, j = tid & 7;
    const int tg = t0 + token, nimg = tg >> 6, s = tg & 63;
    const int idx = sidx[token];
    float ls = 0.f;
    for (int c = j * 32; c < j * 32 + 32; c++) {
      const int ai = ((nimg * 256 + c) * 64) + s;
      const float e = enc[ai];
      const float qv = emb[idx * 256 + c];
      qout[ai] = qv;
      const float d = e - qv;
      ls += d * d;
    }
    redf[tid] = ls;
  }
  __syncthreads();
  for (int st = 128; st > 0; st >>= 1) {
    if (tid < st) redf[tid] += redf[tid + st];
    __syncthreads();
  }
  if (tid == 0) lpart[blockIdx.x] = redf[0];
}

// ---------------------------------------------------------------------------
// Deterministic loss finalize
// ---------------------------------------------------------------------------
__global__ __launch_bounds__(256) void finalize_kernel(const float* __restrict__ vqpart,
                                                       int nvq,
                                                       const float* __restrict__ recpart,
                                                       int nrec,
                                                       float* __restrict__ out4) {
  __shared__ float red[256];
  __shared__ float sums[2];
  float s = 0.f;
  for (int i = threadIdx.x; i < nvq; i += 256) s += vqpart[i];
  red[threadIdx.x] = s;
  __syncthreads();
  for (int st = 128; st > 0; st >>= 1) {
    if (threadIdx.x < st) red[threadIdx.x] += red[threadIdx.x + st];
    __syncthreads();
  }
  if (threadIdx.x == 0) sums[0] = red[0];
  __syncthreads();
  s = 0.f;
  for (int i = threadIdx.x; i < nrec; i += 256) s += recpart[i];
  red[threadIdx.x] = s;
  __syncthreads();
  for (int st = 128; st > 0; st >>= 1) {
    if (threadIdx.x < st) red[threadIdx.x] += red[threadIdx.x + st];
    __syncthreads();
  }
  if (threadIdx.x == 0) {
    sums[1] = red[0];
    const float dict = sums[0] / 4194304.f;    // 16384 tokens * 256 ch
    const float rec  = sums[1] / 786432.f;     // 256*3*32*32
    out4[0] = rec + dict + 1.0f * dict;  // total (BETA=1, commit==dict fwd)
    out4[1] = rec;
    out4[2] = dict;
    out4[3] = dict;
  }
}

// ---------------------------------------------------------------------------
// Host orchestration
// ---------------------------------------------------------------------------
extern "C" void kernel_launch(void* const* d_in, const int* in_sizes, int n_in,
                              void* d_out, int out_size, void* d_ws, size_t ws_size,
                              hipStream_t stream) {
  (void)in_sizes; (void)n_in; (void)out_size; (void)ws_size;
  const float* x      = (const float*)d_in[0];
  const float* enc_w1 = (const float*)d_in[1];
  const float* enc_b1 = (const float*)d_in[2];
  const float* enc_w2 = (const float*)d_in[3];
  const float* enc_b2 = (const float*)d_in[4];
  const float* rms_w  = (const float*)d_in[5];   // (4,2,256,8,8)
  const float* c3w    = (const float*)d_in[6];   // (4,256,256,3,3)
  const float* c3b    = (const float*)d_in[7];   // (4,256)
  const float* c1w    = (const float*)d_in[8];   // (4,256,256,1,1)
  const float* c1b    = (const float*)d_in[9];   // (4,256)
  const float* emb    = (const float*)d_in[10];  // (512,256)
  const float* dec_w1 = (const float*)d_in[11];  // (256,256,4,4) in,out
  const float* dec_b1 = (const float*)d_in[12];
  const float* dec_w2 = (const float*)d_in[13];  // (256,3,4,4)
  const float* dec_b2 = (const float*)d_in[14];

  float* out = (float*)d_out;
  float* decoded = out + 4;                      // (256,3,32,32)

  float* ws   = (float*)d_ws;                    // ~117.5 MB used
  float* big0 = ws;                              // 256*256*16*16
  float* B0   = big0 + 16777216;                 // 256*256*8*8
  float* B1   = B0 + 4194304;
  float* B2   = B1 + 4194304;
  float* enrm = B2 + 4194304;                    // 512
  float* vqp  = enrm + 512;                      // 512 partials
  float* recp = vqp + 512;                       // 2048 partials

  // ---- encoder conv1: x -> big0, relu ----
  {
    GP p{}; p.in = x; p.wgt = enc_w1; p.bias = enc_b1; p.out = big0;
    p.IC = 3; p.IH = 32; p.IW = 32; p.OC = 256; p.OH = 16; p.OW = 16;
    p.KH = 4; p.KW = 4; p.S = 2; p.P = 1; p.M = 65536; p.K = 48;
    p.sOW = 4; p.sOHW = 8;
    gemm_conv<0, 1, 0, 0, 4, 2, 2><<<dim3(1024, 4, 1), 256, 0, stream>>>(p);
  }
  // ---- encoder conv2: big0 -> B0, relu ----
  {
    GP p{}; p.in = big0; p.wgt = enc_w2; p.bias = enc_b2; p.out = B0;
    p.IC = 256; p.IH = 16; p.IW = 16; p.OC = 256; p.OH = 8; p.OW = 8;
    p.KH = 4; p.KW = 4; p.S = 2; p.P = 1; p.M = 16384; p.K = 4096;
    p.sOW = 3; p.sOHW = 6;
    gemm_conv<0, 1, 0, 0, 4, 2, 2><<<dim3(256, 4, 1), 256, 0, stream>>>(p);
  }

  auto res_block = [&](float* X, int rb) {
    rmsnorm_kernel<<<256, 256, 0, stream>>>(X, rms_w + (rb * 2 + 0) * 16384, B2);
    {
      GP p{}; p.in = B2; p.wgt = c3w + rb * 256 * 256 * 9; p.bias = c3b + rb * 256;
      p.resid = X; p.out = X;
      p.IC = 256; p.IH = 8; p.IW = 8; p.OC = 256; p.OH = 8; p.OW = 8;
      p.KH = 3; p.KW = 3; p.S = 1; p.P = 1; p.M = 16384; p.K = 2304;
      p.sOW = 3; p.sOHW = 6;
      gemm_conv<0, 1, 1, 0, 4, 2, 2><<<dim3(256, 4, 1), 256, 0, stream>>>(p);
    }
    rmsnorm_kernel<<<256, 256, 0, stream>>>(X, rms_w + (rb * 2 + 1) * 16384, B2);
    {
      GP p{}; p.in = B2; p.wgt = c1w + rb * 256 * 256; p.bias = c1b + rb * 256;
      p.resid = X; p.out = X;
      p.IC = 256; p.IH = 8; p.IW = 8; p.OC = 256; p.OH = 8; p.OW = 8;
      p.KH = 1; p.KW = 1; p.S = 1; p.P = 0; p.M = 16384; p.K = 256;
      p.sOW = 3; p.sOHW = 6;
      gemm_conv<0, 1, 1, 0, 4, 2, 2><<<dim3(256, 4, 1), 256, 0, stream>>>(p);
    }
  };

  res_block(B0, 0);
  res_block(B0, 1);   // B0 = enc

  // ---- VQ: B0 -> q in B1, dict/commit loss partials ----
  enorm_kernel<<<512, 256, 0, stream>>>(emb, enrm);
  vq_kernel<<<512, 256, 0, stream>>>(B0, emb, enrm, B1, vqp);

  res_block(B1, 2);
  res_block(B1, 3);

  // ---- deconv1: B1 -> big0, relu (4 parity classes) ----
  {
    GP p{}; p.in = B1; p.wgt = dec_w1; p.bias = dec_b1; p.out = big0;
    p.IC = 256; p.IH = 8; p.IW = 8; p.OC = 256; p.OH = 16; p.OW = 16;
    p.KH = 4; p.KW = 4; p.S = 2; p.P = 1; p.M = 16384; p.K = 1024;
    p.sOW = 4; p.sOHW = 8;
    gemm_conv<1, 1, 0, 0, 4, 2, 2><<<dim3(256, 4, 4), 256, 0, stream>>>(p);
  }
  // ---- deconv2: big0 -> decoded + reconstruction loss partials ----
  {
    GP p{}; p.in = big0; p.wgt = dec_w2; p.bias = dec_b2; p.out = decoded;
    p.xref = x; p.lpart = recp;
    p.IC = 256; p.IH = 16; p.IW = 16; p.OC = 3; p.OH = 32; p.OW = 32;
    p.KH = 4; p.KW = 4; p.S = 2; p.P = 1; p.M = 65536; p.K = 1024;
    p.sOW = 5; p.sOHW = 10;
    gemm_conv<1, 0, 0, 1, 8, 1, 1><<<dim3(512, 1, 4), 256, 0, stream>>>(p);
  }

  finalize_kernel<<<1, 256, 0, stream>>>(vqp, 512, recp, 2048, out);
}
